// Block_70832600646212
// MI455X (gfx1250) — matrix operand order repaired and checked by reading, not compiled, not run on hardware
//
#include <hip/hip_runtime.h>

typedef unsigned short u16;
typedef unsigned int u32;
typedef __attribute__((ext_vector_type(16))) __bf16 v16bf;
typedef __attribute__((ext_vector_type(8))) float v8f;

constexpr int kB = 2, kT = 2048, kD = 2048, kN = 8, kKh = 2, kH = 256, kF = 8192;
constexpr int kTok = kB * kT;   // 4096 tokens
constexpr int kNH = kN * kH;    // 2048
constexpr int kKH = kKh * kH;   // 512
constexpr int kWin = 512;
constexpr float kMask = -2.3819763e+38f;
constexpr float kCap = 50.0f;
constexpr float kQScale = 0.0625f;  // 256^-0.5
constexpr float kEps = 1e-6f;

__device__ __forceinline__ u16 f2bf(float f) {
  u32 x = __builtin_bit_cast(u32, f);
  x += 0x7fffu + ((x >> 16) & 1u);
  return (u16)(x >> 16);
}
__device__ __forceinline__ float bf2f(u16 h) {
  u32 x = ((u32)h) << 16;
  return __builtin_bit_cast(float, x);
}

union Frag16 { v16bf v; u32 u[8]; };

__device__ __forceinline__ float block_reduce_sum(float v, float* sbuf) {
  int lane = threadIdx.x & 31, w = threadIdx.x >> 5;
#pragma unroll
  for (int off = 16; off; off >>= 1) v += __shfl_xor(v, off, 32);
  if (lane == 0) sbuf[w] = v;
  __syncthreads();
  int nw = blockDim.x >> 5;
  float r = ((int)threadIdx.x < nw) ? sbuf[threadIdx.x] : 0.0f;
  if (w == 0) {
#pragma unroll
    for (int off = 4; off; off >>= 1) r += __shfl_xor(r, off, 32);
    if (lane == 0) sbuf[0] = r;
  }
  __syncthreads();
  float out = sbuf[0];
  __syncthreads();
  return out;
}

// ---------------- weight prep ----------------
__global__ void cast_bf16_kernel(const float* __restrict__ s, u16* __restrict__ d, int n) {
  for (int i = blockIdx.x * blockDim.x + threadIdx.x; i < n; i += gridDim.x * blockDim.x)
    d[i] = f2bf(s[i]);
}

// q_w [N,D,H] -> [D, N*H]
__global__ void permute_qw_kernel(const float* __restrict__ s, u16* __restrict__ d) {
  const int total = kN * kD * kH;
  for (int i = blockIdx.x * blockDim.x + threadIdx.x; i < total; i += gridDim.x * blockDim.x) {
    int n = i / (kD * kH);
    int r = i - n * (kD * kH);
    int dd = r / kH;
    int h = r - dd * kH;
    d[(size_t)dd * kNH + n * kH + h] = f2bf(s[i]);
  }
}

// kv_w [2,K,D,H] -> [D, 2*K*H]  (cols: c*512 + k*256 + h)
__global__ void permute_kvw_kernel(const float* __restrict__ s, u16* __restrict__ d) {
  const int total = 2 * kKh * kD * kH;
  for (int i = blockIdx.x * blockDim.x + threadIdx.x; i < total; i += gridDim.x * blockDim.x) {
    int c = i / (kKh * kD * kH);
    int r = i - c * (kKh * kD * kH);
    int kk = r / (kD * kH);
    int r2 = r - kk * (kD * kH);
    int dd = r2 / kH;
    int h = r2 - dd * kH;
    d[(size_t)dd * (2 * kKH) + (c * kKh + kk) * kH + h] = f2bf(s[i]);
  }
}

// ---------------- AltUp predict ----------------
__global__ __launch_bounds__(256) void altup_predict_kernel(
    const float* __restrict__ x, const float* __restrict__ router_w,
    const float* __restrict__ pred_coefs, float* __restrict__ pred) {
  __shared__ float red[8];
  const int tok = blockIdx.x, tid = threadIdx.x;
  const float* x0p = x + (size_t)tok * kD;
  float x0[8];
  float dots[4] = {0.f, 0.f, 0.f, 0.f};
#pragma unroll
  for (int j = 0; j < 8; ++j) {
    int d = tid + 256 * j;
    float v = x0p[d];
    x0[j] = v;
    const float* rw = router_w + (size_t)d * 4;
    dots[0] += v * rw[0]; dots[1] += v * rw[1];
    dots[2] += v * rw[2]; dots[3] += v * rw[3];
  }
  float mods[4];
  for (int p = 0; p < 4; ++p)
    mods[p] = tanhf(block_reduce_sum(dots[p], red) * (1.0f / kD));
  float coef[4][4];
#pragma unroll
  for (int i = 0; i < 4; ++i)
#pragma unroll
    for (int j = 0; j < 4; ++j) {
      float c = 0.f;
      for (int p = 0; p < 4; ++p) c += mods[p] * pred_coefs[p * 16 + i * 4 + j];
      coef[i][j] = c;
    }
#pragma unroll
  for (int j = 0; j < 8; ++j) {
    int d = tid + 256 * j;
    float xv[4];
    xv[0] = x0[j];
    for (int i = 1; i < 4; ++i) xv[i] = x[((size_t)i * kTok + tok) * kD + d];
#pragma unroll
    for (int i = 0; i < 4; ++i) {
      float pv = xv[i] + coef[i][0] * xv[0] + coef[i][1] * xv[1] +
                 coef[i][2] * xv[2] + coef[i][3] * xv[3];
      pred[((size_t)i * kTok + tok) * kD + d] = pv;
    }
  }
}

// ---------------- RMS norm -> bf16 ----------------
__global__ __launch_bounds__(256) void rmsnorm_bf16_kernel(
    const float* __restrict__ src, const float* __restrict__ scale, u16* __restrict__ dst) {
  __shared__ float red[8];
  const int tok = blockIdx.x, tid = threadIdx.x;
  const float* row = src + (size_t)tok * kD;
  float v[8], ss = 0.f;
#pragma unroll
  for (int j = 0; j < 8; ++j) { v[j] = row[tid + 256 * j]; ss += v[j] * v[j]; }
  ss = block_reduce_sum(ss, red);
  float rr = rsqrtf(ss * (1.0f / kD) + kEps);
#pragma unroll
  for (int j = 0; j < 8; ++j) {
    int d = tid + 256 * j;
    dst[(size_t)tok * kD + d] = f2bf(v[j] * rr * (1.0f + scale[d]));
  }
}

// gated = pred0 + rmsnorm(attn)*(1+post_scale)
__global__ __launch_bounds__(256) void add_rmsnorm_kernel(
    const float* __restrict__ attn, const float* __restrict__ pred0,
    const float* __restrict__ scale, float* __restrict__ gated) {
  __shared__ float red[8];
  const int tok = blockIdx.x, tid = threadIdx.x;
  const float* row = attn + (size_t)tok * kD;
  float v[8], ss = 0.f;
#pragma unroll
  for (int j = 0; j < 8; ++j) { v[j] = row[tid + 256 * j]; ss += v[j] * v[j]; }
  ss = block_reduce_sum(ss, red);
  float rr = rsqrtf(ss * (1.0f / kD) + kEps);
#pragma unroll
  for (int j = 0; j < 8; ++j) {
    int d = tid + 256 * j;
    gated[(size_t)tok * kD + d] =
        pred0[(size_t)tok * kD + d] + v[j] * rr * (1.0f + scale[d]);
  }
}

// ---------------- WMMA GEMM: C[M,N] = A[M,K] * B[K,N], bf16 in, f32 acc ----------------
// Double-buffered LDS staging via GLOBAL_LOAD_ASYNC_TO_LDS_B128 (ASYNCcnt pipeline).
template <int STORE_BF16>
__global__ __launch_bounds__(256) void gemm_bf16_wmma(
    const u16* __restrict__ Ag, const u16* __restrict__ Bg,
    float* __restrict__ Cf, u16* __restrict__ Cbf, int M, int Nn, int Kd) {
  __shared__ u16 As[2][64][40];    // A tile, row-major [m][k], pad 8
  __shared__ u16 Bs[2][32][136];   // B tile, row-major [k][n], pad 8
  const int tid = threadIdx.x;
  const int lane = tid & 31, w = tid >> 5;
  const int wm = w >> 2, wn = w & 3;
  const int m0 = blockIdx.y * 64;
  const int n0 = blockIdx.x * 128;

  v8f zero = {};
  v8f acc[4];
#pragma unroll
  for (int t = 0; t < 4; ++t) acc[t] = zero;

  const int arow = tid >> 2, akb = (tid & 3) * 8;         // 1 x 16B A chunk / thread
  const int brow = tid >> 3, bnc = (tid & 7) * 16;        // 2 x 16B B chunks / thread
  const int mrow0 = wm * 32 + (lane & 15);
  const int ncol0 = wn * 32 + (lane & 15);
  const int kbaseA = (lane & 16) ? 16 : 0;
  const int kbaseB = (lane & 16) ? 16 : 0;

  // async stage of one (A,B) k-slab into LDS buffer `buf`
  auto stage = [&](int buf, int k0) {
    u32 lA = (u32)(size_t)&As[buf][arow][akb];
    unsigned long long gA =
        (unsigned long long)(Ag + (size_t)(m0 + arow) * Kd + k0 + akb);
    asm volatile("global_load_async_to_lds_b128 %0, %1, off"
                 :: "v"(lA), "v"(gA) : "memory");
    u32 lB0 = (u32)(size_t)&Bs[buf][brow][bnc];
    unsigned long long gB0 =
        (unsigned long long)(Bg + (size_t)(k0 + brow) * Nn + n0 + bnc);
    asm volatile("global_load_async_to_lds_b128 %0, %1, off"
                 :: "v"(lB0), "v"(gB0) : "memory");
    u32 lB1 = (u32)(size_t)&Bs[buf][brow][bnc + 8];
    unsigned long long gB1 =
        (unsigned long long)(Bg + (size_t)(k0 + brow) * Nn + n0 + bnc + 8);
    asm volatile("global_load_async_to_lds_b128 %0, %1, off"
                 :: "v"(lB1), "v"(gB1) : "memory");
  };

  stage(0, 0);
  const int nsteps = Kd >> 5;
  for (int s = 0; s < nsteps; ++s) {
    const int k0 = s << 5;
    const int cur = s & 1;
    const bool pf = (s + 1 < nsteps);
    if (pf) {
      stage(cur ^ 1, k0 + 32);
      // ASYNCcnt is in-order: <=3 outstanding means current slab has landed.
      asm volatile("s_wait_asynccnt 0x3" ::: "memory");
    } else {
      asm volatile("s_wait_asynccnt 0x0" ::: "memory");
    }
    __syncthreads();

    Frag16 fa[2], fb[2];
#pragma unroll
    for (int mi = 0; mi < 2; ++mi) {
      const u16* ar = &As[cur][mrow0 + mi * 16][0];
#pragma unroll
      for (int i = 0; i < 8; ++i) {
        int kloc = kbaseA + 2 * i;
        fa[mi].u[i] = *(const u32*)(ar + kloc);
      }
    }
#pragma unroll
    for (int nj = 0; nj < 2; ++nj) {
      int col = ncol0 + nj * 16;
#pragma unroll
      for (int i = 0; i < 8; ++i) {
        u32 lo = Bs[cur][kbaseB + 2 * i][col];
        u32 hi = Bs[cur][kbaseB + 2 * i + 1][col];
        fb[nj].u[i] = lo | (hi << 16);
      }
    }
#pragma unroll
    for (int mi = 0; mi < 2; ++mi)
#pragma unroll
      for (int nj = 0; nj < 2; ++nj)
        acc[mi * 2 + nj] = __builtin_amdgcn_wmma_f32_16x16x32_bf16(
            false, fa[mi].v, false, fb[nj].v, (short)0, acc[mi * 2 + nj], false, false);
    __syncthreads();
  }

  const int rbase = (lane & 16) ? 8 : 0;
#pragma unroll
  for (int mi = 0; mi < 2; ++mi)
#pragma unroll
    for (int nj = 0; nj < 2; ++nj) {
      v8f c = acc[mi * 2 + nj];
      int col = n0 + wn * 32 + nj * 16 + (lane & 15);
#pragma unroll
      for (int i = 0; i < 8; ++i) {
        int r = m0 + wm * 32 + mi * 16 + rbase + i;
        if (STORE_BF16)
          Cbf[(size_t)r * Nn + col] = f2bf(c[i]);
        else
          Cf[(size_t)r * Nn + col] = c[i];
      }
    }
}

// ---------------- QK-norm + RoPE ----------------
__global__ __launch_bounds__(256) void rope_q_kernel(
    const float* __restrict__ qf, const int* __restrict__ segpos,
    const float* __restrict__ qs, u16* __restrict__ qbf) {
  const int tok = blockIdx.x;
  const int wave = threadIdx.x >> 5, lane = threadIdx.x & 31;
  const float* row = qf + (size_t)tok * kNH + wave * kH;
  float x1[4], x2[4], ss = 0.f;
#pragma unroll
  for (int jj = 0; jj < 4; ++jj) {
    int pr = lane + 32 * jj;
    x1[jj] = row[pr];
    x2[jj] = row[pr + 128];
    ss += x1[jj] * x1[jj] + x2[jj] * x2[jj];
  }
#pragma unroll
  for (int off = 16; off; off >>= 1) ss += __shfl_xor(ss, off, 32);
  float rr = rsqrtf(ss * (1.0f / kH) + kEps);
  float pos = (float)segpos[tok];
  u16* orow = qbf + (size_t)tok * kNH + wave * kH;
#pragma unroll
  for (int jj = 0; jj < 4; ++jj) {
    int pr = lane + 32 * jj;
    float ts = __powf(10000.0f, pr * (1.0f / 128.0f));
    float ang = pos / ts;
    float s = __sinf(ang), c = __cosf(ang);
    float n1 = x1[jj] * rr * (1.0f + qs[pr]);
    float n2 = x2[jj] * rr * (1.0f + qs[pr + 128]);
    orow[pr] = f2bf((n1 * c - n2 * s) * kQScale);
    orow[pr + 128] = f2bf((n2 * c + n1 * s) * kQScale);
  }
}

// waves 0..1: K heads (norm+scale+rope -> kbf [tok, K*H]); waves 2..3: V heads (norm -> vT [b,kh,h,t])
__global__ __launch_bounds__(128) void rope_kv_kernel(
    const float* __restrict__ kvf, const int* __restrict__ segpos,
    const float* __restrict__ ks, u16* __restrict__ kbf, u16* __restrict__ vT) {
  const int tok = blockIdx.x;
  const int wave = threadIdx.x >> 5, lane = threadIdx.x & 31;
  const int is_v = wave >> 1, kh = wave & 1;
  const float* row = kvf + (size_t)tok * (2 * kKH) + is_v * kKH + kh * kH;
  if (!is_v) {
    float x1[4], x2[4], ss = 0.f;
#pragma unroll
    for (int jj = 0; jj < 4; ++jj) {
      int pr = lane + 32 * jj;
      x1[jj] = row[pr];
      x2[jj] = row[pr + 128];
      ss += x1[jj] * x1[jj] + x2[jj] * x2[jj];
    }
#pragma unroll
    for (int off = 16; off; off >>= 1) ss += __shfl_xor(ss, off, 32);
    float rr = rsqrtf(ss * (1.0f / kH) + kEps);
    float pos = (float)segpos[tok];
    u16* orow = kbf + (size_t)tok * kKH + kh * kH;
#pragma unroll
    for (int jj = 0; jj < 4; ++jj) {
      int pr = lane + 32 * jj;
      float ts = __powf(10000.0f, pr * (1.0f / 128.0f));
      float ang = pos / ts;
      float s = __sinf(ang), c = __cosf(ang);
      float n1 = x1[jj] * rr * (1.0f + ks[pr]);
      float n2 = x2[jj] * rr * (1.0f + ks[pr + 128]);
      orow[pr] = f2bf(n1 * c - n2 * s);
      orow[pr + 128] = f2bf(n2 * c + n1 * s);
    }
  } else {
    float v[8], ss = 0.f;
#pragma unroll
    for (int jj = 0; jj < 8; ++jj) {
      v[jj] = row[lane + 32 * jj];
      ss += v[jj] * v[jj];
    }
#pragma unroll
    for (int off = 16; off; off >>= 1) ss += __shfl_xor(ss, off, 32);
    float rr = rsqrtf(ss * (1.0f / kH) + kEps);
    const int b = tok >> 11, tl = tok & (kT - 1);
#pragma unroll
    for (int jj = 0; jj < 8; ++jj) {
      int h = lane + 32 * jj;
      vT[((size_t)((b * kKh + kh) * kH + h)) * kT + tl] = f2bf(v[jj] * rr);
    }
  }
}

// ---------------- sliding-window flash attention, one wave per 16-query tile per head ----------------
__global__ __launch_bounds__(32) void fa_sliding_wmma(
    const u16* __restrict__ qb, const u16* __restrict__ kbf,
    const u16* __restrict__ vT, const int* __restrict__ segpos,
    u16* __restrict__ enc) {
  __shared__ u16 Pl[16][32];
  const int lane = threadIdx.x;
  const int tile = blockIdx.x;
  const int n = blockIdx.y;
  const int row0 = tile * 16;
  const int b = row0 / kT;
  const int t0 = row0 - b * kT;
  const int kh = n >> 2;  // GQA group size = 4

  const int m = lane & 15;
  const int khalf = (lane >> 4) & 1;
  const int kbaseA = khalf ? 16 : 0;
  const int kbaseB = khalf ? 16 : 0;

  // Q A-fragments: 8 K-chunks of 32 over H=256
  Frag16 aq[8];
  {
    const u16* qrow = qb + (size_t)(row0 + m) * kNH + n * kH;
#pragma unroll
    for (int c = 0; c < 8; ++c)
#pragma unroll
      for (int i = 0; i < 8; ++i) {
        int kloc = kbaseA + 2 * i;
        aq[c].u[i] = *(const u32*)(qrow + c * 32 + kloc);
      }
  }

  int post[8];
#pragma unroll
  for (int i = 0; i < 8; ++i) post[i] = segpos[row0 + khalf * 8 + i];

  float rmax[8], rsum[8];
  v8f zero = {};
  v8f accO[16];
#pragma unroll
  for (int i = 0; i < 8; ++i) { rmax[i] = kMask; rsum[i] = 0.f; }
#pragma unroll
  for (int t = 0; t < 16; ++t) accO[t] = zero;

  int smin = t0 - (kWin - 1);
  if (smin < 0) smin = 0;
  smin &= ~31;
  const int send = t0 + 15;

  for (int s0 = smin; s0 <= send; s0 += 32) {
    v8f sh[2];
#pragma unroll
    for (int h2 = 0; h2 < 2; ++h2) {
      v8f s = zero;
      const int key_local = s0 + h2 * 16 + (lane & 15);
      const int ktok = b * kT + key_local;
      const u16* krow = kbf + (size_t)ktok * kKH + kh * kH;
#pragma unroll
      for (int c = 0; c < 8; ++c) {
        Frag16 fb;
#pragma unroll
        for (int i = 0; i < 8; ++i)
          fb.u[i] = *(const u32*)(krow + c * 32 + kbaseB + 2 * i);
        s = __builtin_amdgcn_wmma_f32_16x16x32_bf16(false, aq[c].v, false, fb.v,
                                                    (short)0, s, false, false);
      }
      const int pos_s = segpos[ktok];
#pragma unroll
      for (int i = 0; i < 8; ++i) {
        float sv = kCap * tanhf(s[i] * (1.0f / kCap));
        bool ok = (pos_s <= post[i]) && (pos_s > post[i] - kWin);
        s[i] = ok ? sv : kMask;
      }
      sh[h2] = s;
    }
    // online softmax update
#pragma unroll
    for (int i = 0; i < 8; ++i) {
      float mx = fmaxf(sh[0][i], sh[1][i]);
#pragma unroll
      for (int off = 8; off; off >>= 1) mx = fmaxf(mx, __shfl_xor(mx, off, 32));
      float nm = fmaxf(rmax[i], mx);
      float sc = __expf(rmax[i] - nm);
      rmax[i] = nm;
      float p0 = __expf(sh[0][i] - nm);
      float p1 = __expf(sh[1][i] - nm);
      sh[0][i] = p0;
      sh[1][i] = p1;
      float ps = p0 + p1;
#pragma unroll
      for (int off = 8; off; off >>= 1) ps += __shfl_xor(ps, off, 32);
      rsum[i] = rsum[i] * sc + ps;
#pragma unroll
      for (int t = 0; t < 16; ++t) accO[t][i] *= sc;
    }
    // P -> LDS (bf16), reload in A-fragment layout
#pragma unroll
    for (int h2 = 0; h2 < 2; ++h2)
#pragma unroll
      for (int i = 0; i < 8; ++i)
        Pl[khalf * 8 + i][h2 * 16 + (lane & 15)] = f2bf(sh[h2][i]);
    asm volatile("s_wait_dscnt 0" ::: "memory");
    Frag16 pf;
#pragma unroll
    for (int i = 0; i < 8; ++i) {
      int kloc = kbaseA + 2 * i;
      pf.u[i] = *(const u32*)(&Pl[m][0] + kloc);
    }
    // O += P x V  (V pre-transposed: rows = h, cols = t)
    const u16* vbase = vT + (size_t)((b * kKh + kh) * kH) * kT;
#pragma unroll
    for (int hh = 0; hh < 16; ++hh) {
      const u16* vcol = vbase + (size_t)(hh * 16 + (lane & 15)) * kT;
      Frag16 fv;
#pragma unroll
      for (int i = 0; i < 8; ++i)
        fv.u[i] = *(const u32*)(vcol + s0 + kbaseB + 2 * i);
      accO[hh] = __builtin_amdgcn_wmma_f32_16x16x32_bf16(false, pf.v, false, fv.v,
                                                         (short)0, accO[hh], false, false);
    }
  }
  float inv[8];
#pragma unroll
  for (int i = 0; i < 8; ++i) inv[i] = 1.0f / rsum[i];
#pragma unroll
  for (int hh = 0; hh < 16; ++hh)
#pragma unroll
    for (int i = 0; i < 8; ++i) {
      int r = row0 + khalf * 8 + i;
      int col = hh * 16 + (lane & 15);
      enc[(size_t)r * kNH + n * kH + col] = f2bf(accO[hh][i] * inv[i]);
    }
}

// ---------------- gated GELU ----------------
__global__ void gelu_mul_kernel(const u16* __restrict__ g, const u16* __restrict__ u,
                                u16* __restrict__ o, int n) {
  for (int i = blockIdx.x * blockDim.x + threadIdx.x; i < n; i += gridDim.x * blockDim.x) {
    float x = bf2f(g[i]);
    float t = tanhf(0.7978845608f * (x + 0.044715f * x * x * x));
    float ge = 0.5f * x * (1.0f + t);
    o[i] = f2bf(ge * bf2f(u[i]));
  }
}

// ---------------- final: post-ffw norm + residual + AltUp correct ----------------
__global__ __launch_bounds__(256) void final_correct_kernel(
    const float* __restrict__ ffw, const float* __restrict__ gated,
    const float* __restrict__ post_ffw_scale, const float* __restrict__ cos_scale,
    const float* __restrict__ router_w, const float* __restrict__ corr_coefs,
    const float* __restrict__ pred, float* __restrict__ out) {
  __shared__ float red[8];
  const int tok = blockIdx.x, tid = threadIdx.x;
  float f[8], ss = 0.f;
#pragma unroll
  for (int j = 0; j < 8; ++j) {
    f[j] = ffw[(size_t)tok * kD + tid + 256 * j];
    ss += f[j] * f[j];
  }
  float rr = rsqrtf(block_reduce_sum(ss, red) * (1.0f / kD) + kEps);
  float act[8];
  float dots[4] = {0.f, 0.f, 0.f, 0.f};
#pragma unroll
  for (int j = 0; j < 8; ++j) {
    int d = tid + 256 * j;
    float a = (gated[(size_t)tok * kD + d] + f[j] * rr * (1.0f + post_ffw_scale[d])) * cos_scale[d];
    act[j] = a;
    const float* rw = router_w + (size_t)d * 4;
    dots[0] += a * rw[0]; dots[1] += a * rw[1];
    dots[2] += a * rw[2]; dots[3] += a * rw[3];
  }
  float mods[4];
  for (int p = 0; p < 4; ++p)
    mods[p] = tanhf(block_reduce_sum(dots[p], red) * (1.0f / kD));
  float cc[4];
#pragma unroll
  for (int i = 0; i < 4; ++i) {
    float c = 1.0f;
    for (int p = 0; p < 4; ++p) c += mods[p] * corr_coefs[p * 4 + i];
    cc[i] = c;
  }
#pragma unroll
  for (int j = 0; j < 8; ++j) {
    int d = tid + 256 * j;
    float inn = act[j] - pred[(size_t)tok * kD + d];
#pragma unroll
    for (int i = 0; i < 4; ++i)
      out[((size_t)i * kTok + tok) * kD + d] =
          pred[((size_t)i * kTok + tok) * kD + d] + cc[i] * inn;
  }
}

// ---------------- host side ----------------
extern "C" void kernel_launch(void* const* d_in, const int* in_sizes, int n_in,
                              void* d_out, int out_size, void* d_ws, size_t ws_size,
                              hipStream_t stream) {
  const float* x = (const float*)d_in[0];
  const int* segpos = (const int*)d_in[1];
  // d_in[2] attn_mask: equals causal(pos) -> recomputed from segpos, unused
  const float* router_w = (const float*)d_in[3];
  const float* pred_coefs = (const float*)d_in[4];
  const float* corr_coefs = (const float*)d_in[5];
  const float* cos_scale = (const float*)d_in[6];
  const float* pre_attn_scale = (const float*)d_in[7];
  const float* q_w = (const float*)d_in[8];
  const float* kv_w = (const float*)d_in[9];
  const float* qk_q_scale = (const float*)d_in[10];
  const float* qk_k_scale = (const float*)d_in[11];
  const float* o_w = (const float*)d_in[12];
  const float* post_attn_scale = (const float*)d_in[13];
  const float* pre_ffw_scale = (const float*)d_in[14];
  const float* gate_w = (const float*)d_in[15];
  const float* up_w = (const float*)d_in[16];
  const float* down_w = (const float*)d_in[17];
  const float* post_ffw_scale = (const float*)d_in[18];
  float* out = (float*)d_out;

  size_t off = 0;
  auto alloc = [&](size_t bytes) -> void* {
    void* p = (char*)d_ws + off;
    off += (bytes + 255) & ~(size_t)255;
    return p;
  };
  float* pred   = (float*)alloc((size_t)4 * kTok * kD * 4);
  u16* attn_in  = (u16*)alloc((size_t)kTok * kD * 2);
  u16* wq       = (u16*)alloc((size_t)kD * kNH * 2);
  u16* wkv      = (u16*)alloc((size_t)kD * (2 * kKH) * 2);
  u16* wo       = (u16*)alloc((size_t)kNH * kD * 2);
  u16* wg       = (u16*)alloc((size_t)kD * kF * 2);
  u16* wu       = (u16*)alloc((size_t)kD * kF * 2);
  u16* wd       = (u16*)alloc((size_t)kF * kD * 2);
  float* qf     = (float*)alloc((size_t)kTok * kNH * 4);
  float* kvf    = (float*)alloc((size_t)kTok * 2 * kKH * 4);
  u16* qbf      = (u16*)alloc((size_t)kTok * kNH * 2);
  u16* kbf      = (u16*)alloc((size_t)kTok * kKH * 2);
  u16* vT       = (u16*)alloc((size_t)kB * kKh * kH * kT * 2);
  u16* encb     = (u16*)alloc((size_t)kTok * kNH * 2);
  float* attnf  = (float*)alloc((size_t)kTok * kD * 4);
  float* gatedf = (float*)alloc((size_t)kTok * kD * 4);
  u16* hbf      = (u16*)alloc((size_t)kTok * kD * 2);
  u16* gbf      = (u16*)alloc((size_t)kTok * kF * 2);
  u16* ubf      = (u16*)alloc((size_t)kTok * kF * 2);
  u16* abf      = (u16*)alloc((size_t)kTok * kF * 2);
  float* ffwf   = (float*)alloc((size_t)kTok * kD * 4);

  // weight prep (bf16 GEMM layouts)
  permute_qw_kernel<<<2048, 256, 0, stream>>>(q_w, wq);
  permute_kvw_kernel<<<1024, 256, 0, stream>>>(kv_w, wkv);
  cast_bf16_kernel<<<2048, 256, 0, stream>>>(o_w, wo, kNH * kD);
  cast_bf16_kernel<<<8192, 256, 0, stream>>>(gate_w, wg, kD * kF);
  cast_bf16_kernel<<<8192, 256, 0, stream>>>(up_w, wu, kD * kF);
  cast_bf16_kernel<<<8192, 256, 0, stream>>>(down_w, wd, kF * kD);

  // AltUp predict
  altup_predict_kernel<<<kTok, 256, 0, stream>>>(x, router_w, pred_coefs, pred);
  // pre-attn norm (on predictions[0])
  rmsnorm_bf16_kernel<<<kTok, 256, 0, stream>>>(pred, pre_attn_scale, attn_in);

  // QKV projections
  gemm_bf16_wmma<0><<<dim3(kNH / 128, kTok / 64), 256, 0, stream>>>(
      attn_in, wq, qf, nullptr, kTok, kNH, kD);
  gemm_bf16_wmma<0><<<dim3((2 * kKH) / 128, kTok / 64), 256, 0, stream>>>(
      attn_in, wkv, kvf, nullptr, kTok, 2 * kKH, kD);

  // QK norm + RoPE, V norm + transpose
  rope_q_kernel<<<kTok, 256, 0, stream>>>(qf, segpos, qk_q_scale, qbf);
  rope_kv_kernel<<<kTok, 128, 0, stream>>>(kvf, segpos, qk_k_scale, kbf, vT);

  // sliding-window attention
  fa_sliding_wmma<<<dim3(kTok / 16, kN), 32, 0, stream>>>(qbf, kbf, vT, segpos, encb);

  // O projection
  gemm_bf16_wmma<0><<<dim3(kD / 128, kTok / 64), 256, 0, stream>>>(
      encb, wo, attnf, nullptr, kTok, kD, kNH);

  // post-attn norm + residual
  add_rmsnorm_kernel<<<kTok, 256, 0, stream>>>(attnf, pred, post_attn_scale, gatedf);
  rmsnorm_bf16_kernel<<<kTok, 256, 0, stream>>>(gatedf, pre_ffw_scale, hbf);

  // FFN
  gemm_bf16_wmma<1><<<dim3(kF / 128, kTok / 64), 256, 0, stream>>>(
      hbf, wg, nullptr, gbf, kTok, kF, kD);
  gemm_bf16_wmma<1><<<dim3(kF / 128, kTok / 64), 256, 0, stream>>>(
      hbf, wu, nullptr, ubf, kTok, kF, kD);
  gelu_mul_kernel<<<8192, 256, 0, stream>>>(gbf, ubf, abf, kTok * kF);
  gemm_bf16_wmma<0><<<dim3(kD / 128, kTok / 64), 256, 0, stream>>>(
      abf, wd, ffwf, nullptr, kTok, kD, kF);

  // final norm + residual + AltUp correct -> d_out
  final_correct_kernel<<<kTok, 256, 0, stream>>>(
      ffwf, gatedf, post_ffw_scale, cos_scale, router_w, corr_coefs, pred, out);

  (void)in_sizes; (void)n_in; (void)out_size; (void)ws_size;
}
